// Attend_55825984913997
// MI455X (gfx1250) — matrix operand order repaired and checked by reading, not compiled, not run on hardware
//
#include <hip/hip_runtime.h>
#include <hip/hip_bf16.h>

// ---------------------------------------------------------------------------
// Flash-attention (causal + additive bias + key padding mask) for MI455X
// gfx1250: wave32, v_wmma_f32_16x16x32_bf16 for both GEMMs, f32 accum.
// Block = 128 threads (4 waves). Each block: one (b,h), 64 query rows.
// Each wave: a 16-row query tile. Keys processed in blocks of 32 with
// ping-pong (double-buffered) K/V LDS staging: one barrier per iteration.
// ---------------------------------------------------------------------------

typedef __attribute__((ext_vector_type(16))) __bf16 v16bf;
typedef __attribute__((ext_vector_type(8)))  float  v8f;

union ABf16 {
    v16bf v;
    uint4 q[2];
    unsigned short u[16];
    unsigned int w[8];
};

static __device__ __forceinline__ unsigned int f2bf(float f) {
    union { float f; unsigned int u; } x;
    x.f = f;
    unsigned int u = x.u;
    // round-to-nearest-even bf16
    unsigned int r = u + 0x7FFFu + ((u >> 16) & 1u);
    return r >> 16;
}

static __device__ __forceinline__ unsigned int pk2bf(float lo, float hi) {
    return f2bf(lo) | (f2bf(hi) << 16);
}

static __device__ __forceinline__ v8f wmma_bf16(v16bf a, v16bf b, v8f c) {
    // (neg_a, A, neg_b, B, c_mod, C, reuse_a, reuse_b)
    return __builtin_amdgcn_wmma_f32_16x16x32_bf16(false, a, false, b,
                                                   (short)0, c, false, false);
}

#define ATTN_B 2
#define ATTN_H 16
#define ATTN_S 2048
#define ATTN_D 128
#define ATTN_SCALE 0.08838834764831845f      /* D^-0.5 */
#define ATTN_MASKVAL -1.7014118346046923e38f /* finfo(f32).min / 2 */

__global__ __launch_bounds__(128)
void attend_flash_wmma(const float* __restrict__ Qg,
                       const float* __restrict__ Kg,
                       const float* __restrict__ Vg,
                       const unsigned char* __restrict__ maskg, // [B,1,1,S]
                       const float* __restrict__ biasg,         // [H,S,S]
                       const int* __restrict__ poffset,
                       float* __restrict__ Og) {
    constexpr int S = ATTN_S;
    constexpr int D = ATTN_D;
    constexpr int H = ATTN_H;

    // Ping-pong K (key-major bf16) and V (d-major / transposed bf16) tiles,
    // plus per-wave P re-layout scratch.
    __shared__ alignas(16) unsigned short Klds[2][32 * 128]; // [buf][key][d]
    __shared__ alignas(16) unsigned short Vlds[2][128 * 32]; // [buf][d][key]
    __shared__ alignas(16) unsigned short Plds[4 * 16 * 32]; // [wave][row][col]

    const int tid   = threadIdx.x;       // 0..127
    const int wave  = tid >> 5;          // 0..3
    const int lane  = tid & 31;          // 0..31
    const int lhalf = lane >> 4;         // 0 | 1
    const int l16   = lane & 15;         // 0..15

    const int bh = blockIdx.y;           // 0..31
    const int b  = bh / H;
    const int h  = bh % H;
    const int q0 = blockIdx.x * 64;      // first query row of this block
    const int qrow0 = q0 + wave * 16;    // first query row of this wave

    const int off = poffset[0];          // causal offset (0 in reference)
    const unsigned char* kmask = maskg + (size_t)b * S;

    // Number of 32-key blocks needed (causal upper bound for this block).
    const int nkb = q0 / 32 + 2;

    // ---- Cooperative K/V staging: f32 global -> bf16 LDS ------------------
    auto stage = [&](int skb, int buf) {
        const int key  = tid >> 2;         // 0..31
        const int dseg = (tid & 3) * 32;   // 0,32,64,96
        const size_t gk =
            ((size_t)bh * S + (size_t)(skb * 32 + key)) * D + dseg;
        const float* kp = Kg + gk;
        const float* vp = Vg + gk;
        unsigned int* krow =
            reinterpret_cast<unsigned int*>(&Klds[buf][key * 128 + dseg]);
#pragma unroll
        for (int i = 0; i < 16; ++i)            // packed dword stores
            krow[i] = pk2bf(kp[2 * i], kp[2 * i + 1]);
#pragma unroll
        for (int i = 0; i < 32; ++i)            // transposed scatter
            Vlds[buf][(dseg + i) * 32 + key] = (unsigned short)f2bf(vp[i]);
        if (skb + 1 < nkb) {
            __builtin_prefetch(kp + (size_t)32 * D, 0, 0); // global_prefetch_b8
            __builtin_prefetch(vp + (size_t)32 * D, 0, 0);
        }
    };

    // ---- Load Q tile (16 x 128) into A-layout bf16 registers -------------
    // A operand, contiguous k order used here: lane l16 holds row M=l16; lanes 0-15 get K
    // {base..base+15} with base = c*32; lanes 16-31 same
    // rows, base shifted by 16.
    ABf16 qa[4];
    {
        const float* qp = Qg + ((size_t)bh * S + (size_t)(qrow0 + l16)) * D;
#pragma unroll
        for (int c = 0; c < 4; ++c) {
            const int base = c * 32 + lhalf * 16;
#pragma unroll
            for (int i = 0; i < 4; ++i)
                qa[c].w[i] = pk2bf(qp[base + 2 * i], qp[base + 2 * i + 1]);
#pragma unroll
            for (int i = 0; i < 4; ++i)
                qa[c].w[4 + i] =
                    pk2bf(qp[base + 8 + 2 * i], qp[base + 8 + 2 * i + 1]);
        }
    }

    // ---- Accumulators / softmax state ------------------------------------
    v8f acc[8];                      // 8 D-tiles of 16 cols, f32 C layout
#pragma unroll
    for (int dt = 0; dt < 8; ++dt) acc[dt] = {};
    float mrow[8], lrow[8];
#pragma unroll
    for (int r = 0; r < 8; ++r) { mrow[r] = -3.0e38f; lrow[r] = 0.0f; }

    stage(0, 0); // prime the pipeline

    for (int kb = 0; kb < nkb; ++kb) {
        __syncthreads(); // staging for block kb is complete
        const int buf = kb & 1;
        const unsigned short* Kb = Klds[buf];
        const unsigned short* Vb = Vlds[buf];

        // ---- S = Q * K^T for two 16-key halves (f32 C tiles) -------------
        v8f s0 = {};
        v8f s1 = {};
#pragma unroll
        for (int c = 0; c < 4; ++c) {
            // B operand 32x16: lane l holds column (key) l16, contraction
            // rows lhalf*16..+15 -> contiguous 16 bf16 in Klds.
            ABf16 bk0, bk1;
            const unsigned short* p0 = &Kb[(l16) * 128 + c * 32 + lhalf * 16];
            const unsigned short* p1 =
                &Kb[(16 + l16) * 128 + c * 32 + lhalf * 16];
            bk0.q[0] = *reinterpret_cast<const uint4*>(p0);
            bk0.q[1] = *reinterpret_cast<const uint4*>(p0 + 8);
            bk1.q[0] = *reinterpret_cast<const uint4*>(p1);
            bk1.q[1] = *reinterpret_cast<const uint4*>(p1 + 8);
            s0 = wmma_bf16(qa[c].v, bk0.v, s0);
            s1 = wmma_bf16(qa[c].v, bk1.v, s1);
        }

        // ---- bias + masks + online softmax -------------------------------
        const int colBase = kb * 32;
        const int col0 = colBase + l16;
        const int col1 = col0 + 16;
        const int km0 = kmask[col0];
        const int km1 = kmask[col1];
        float alpha[8];
#pragma unroll
        for (int r = 0; r < 8; ++r) {
            const int row = qrow0 + r + lhalf * 8;
            const float* bp =
                biasg + ((size_t)h * S + (size_t)row) * S + colBase;
            const float bb0 = bp[l16];
            const float bb1 = bp[16 + l16];
            const bool v0 = km0 && (col0 <= row - off);
            const bool v1 = km1 && (col1 <= row - off);
            float x0 = v0 ? s0[r] * ATTN_SCALE + bb0 : ATTN_MASKVAL;
            float x1 = v1 ? s1[r] * ATTN_SCALE + bb1 : ATTN_MASKVAL;

            float t = fmaxf(x0, x1);
            t = fmaxf(t, __shfl_xor(t, 1, 16));
            t = fmaxf(t, __shfl_xor(t, 2, 16));
            t = fmaxf(t, __shfl_xor(t, 4, 16));
            t = fmaxf(t, __shfl_xor(t, 8, 16));

            const float mnew = fmaxf(mrow[r], t);
            const float a = __expf(mrow[r] - mnew);
            mrow[r] = mnew;
            alpha[r] = a;

            const float p0 = __expf(x0 - mnew);
            const float p1 = __expf(x1 - mnew);
            float ps = p0 + p1;
            ps += __shfl_xor(ps, 1, 16);
            ps += __shfl_xor(ps, 2, 16);
            ps += __shfl_xor(ps, 4, 16);
            ps += __shfl_xor(ps, 8, 16);
            lrow[r] = lrow[r] * a + ps;

            // C layout -> LDS (row = r + lhalf*8, cols l16 and 16+l16)
            const int prow = r + lhalf * 8;
            Plds[(wave * 16 + prow) * 32 + l16]      = (unsigned short)f2bf(p0);
            Plds[(wave * 16 + prow) * 32 + 16 + l16] = (unsigned short)f2bf(p1);
        }

        // rescale running output accumulators
#pragma unroll
        for (int dt = 0; dt < 8; ++dt)
#pragma unroll
            for (int r = 0; r < 8; ++r)
                acc[dt][r] *= alpha[r];

        // same-wave LDS RAW: DS ops in-order per wave; keep compiler honest
        asm volatile("s_wait_dscnt 0x0" ::: "memory");

        // ---- Reload P in A layout (16x32 bf16) ---------------------------
        ABf16 pa;
        {
            const unsigned short* pp =
                &Plds[(wave * 16 + l16) * 32 + lhalf * 16];
            pa.q[0] = *reinterpret_cast<const uint4*>(pp);       // cols +0..7
            pa.q[1] = *reinterpret_cast<const uint4*>(pp + 8);   // cols +8..15
        }

        // ---- O += P * V --------------------------------------------------
#pragma unroll
        for (int dt = 0; dt < 8; ++dt) {
            ABf16 bv;
            const unsigned short* vp2 = &Vb[(dt * 16 + l16) * 32 + lhalf * 16];
            bv.q[0] = *reinterpret_cast<const uint4*>(vp2);
            bv.q[1] = *reinterpret_cast<const uint4*>(vp2 + 8);
            acc[dt] = wmma_bf16(pa.v, bv.v, acc[dt]);
        }

        // ---- Stage next key block into the other buffer ------------------
        if (kb + 1 < nkb) stage(kb + 1, (kb + 1) & 1);
    }

    // ---- Normalize and write out (C layout -> row-major global) ----------
    float invl[8];
#pragma unroll
    for (int r = 0; r < 8; ++r) invl[r] = 1.0f / lrow[r];

#pragma unroll
    for (int dt = 0; dt < 8; ++dt) {
#pragma unroll
        for (int r = 0; r < 8; ++r) {
            const int row = qrow0 + r + lhalf * 8;
            Og[((size_t)bh * S + (size_t)row) * D + dt * 16 + l16] =
                acc[dt][r] * invl[r];
        }
    }
}

extern "C" void kernel_launch(void* const* d_in, const int* in_sizes, int n_in,
                              void* d_out, int out_size, void* d_ws, size_t ws_size,
                              hipStream_t stream) {
    (void)in_sizes; (void)n_in; (void)out_size; (void)d_ws; (void)ws_size;
    const float*         q    = (const float*)d_in[0];
    const float*         k    = (const float*)d_in[1];
    const float*         v    = (const float*)d_in[2];
    const unsigned char* mask = (const unsigned char*)d_in[3]; // bool [B,1,1,S]
    const float*         bias = (const float*)d_in[4];         // [H,S,S]
    const int*           off  = (const int*)d_in[5];
    float*               out  = (float*)d_out;

    dim3 grid(ATTN_S / 64, ATTN_B * ATTN_H); // (32, 32)
    dim3 block(128);                         // 4 wave32
    attend_flash_wmma<<<grid, block, 0, stream>>>(q, k, v, mask, bias, off, out);
}
